// MambaBlock_52304111731204
// MI455X (gfx1250) — matrix operand order repaired and checked by reading, not compiled, not run on hardware
//
#include <hip/hip_runtime.h>

#define D_MODEL 1024
#define D_STATE 64
#define D_CONV  4
#define D_INNER 2048
#define BATCH   2
#define SEQ_T   1024
#define NTOK    (BATCH * SEQ_T)   /* 2048 tokens */
#define LN_EPS  1e-5f

typedef __attribute__((ext_vector_type(16))) __bf16        v16bf;
typedef __attribute__((ext_vector_type(8)))  float         v8f;
typedef __attribute__((ext_vector_type(8)))  unsigned short us8;
typedef __attribute__((ext_vector_type(4)))  unsigned int  u32x4;
typedef __attribute__((ext_vector_type(8)))  unsigned int  u32x8;

union BfFrag { v16bf v; us8 h[2]; unsigned short s[16]; };

__device__ __forceinline__ unsigned short f2bf(float f) {
  unsigned int u = __float_as_uint(f);
  u += 0x7FFFu + ((u >> 16) & 1u);           // round-to-nearest-even
  return (unsigned short)(u >> 16);
}
__device__ __forceinline__ float sigmoidf_(float x) { return 1.0f / (1.0f + __expf(-x)); }
__device__ __forceinline__ float siluf_(float x)    { return x * sigmoidf_(x); }
__device__ __forceinline__ float softplusf_(float x){ return (x > 20.0f) ? x : log1pf(__expf(x)); }

// ---------------------------------------------------------------------------
// TDM: issue a 2D bf16 tile load (tile_w x tile_h, row stride = stride_elems)
// from global into LDS at byte offset lds_off. D# per ISA 08 §8; 2D tensor ->
// 2-SGPR-group form (VADDR2/VADDR3 = NULL). Tracked with TENSORcnt, in-order
// completion per wave.
// ---------------------------------------------------------------------------
__device__ __forceinline__ void tdm_load_tile_bf16(
    unsigned lds_off, const unsigned short* gsrc,
    unsigned tile_w, unsigned tile_h, unsigned stride_elems)
{
  unsigned long long ga = (unsigned long long)(uintptr_t)gsrc;
  u32x4 g0;
  g0[0] = 1u;                                  // count=1, user descriptor
  g0[1] = lds_off;                             // lds_addr (bytes)
  g0[2] = (unsigned)ga;                        // global_addr[31:0]
  g0[3] = (unsigned)(ga >> 32) | (2u << 30);   // global_addr[56:32] | type=2
  u32x8 g1;
  g1[0] = 1u << 16;                            // data_size=1 (2 bytes), mask=0
  g1[1] = tile_w << 16;                        // tensor_dim0[15:0] @ bits 63:48
  g1[2] = tile_h << 16;                        // tensor_dim1[15:0] @ bits 95:80
  g1[3] = tile_w << 16;                        // tile_dim0 @ bits 127:112
  g1[4] = tile_h;                              // tile_dim1; tile_dim2=0
  g1[5] = stride_elems;                        // tensor_dim0_stride[31:0]
  g1[6] = 0u;
  g1[7] = 0u;
  asm volatile("tensor_load_to_lds %0, %1" :: "s"(g0), "s"(g1) : "memory");
}

// ---------------------------------------------------------------------------
// Weight fp32 [K,N] -> bf16 transposed [N,K]
// ---------------------------------------------------------------------------
__global__ __launch_bounds__(256) void cvt_transpose_kernel(
    const float* __restrict__ src, unsigned short* __restrict__ dst, int K, int N)
{
  size_t gid = (size_t)blockIdx.x * 256 + threadIdx.x;
  if (gid >= (size_t)K * N) return;
  int k = (int)(gid % K);
  int n = (int)(gid / K);
  dst[gid] = f2bf(src[(size_t)k * N + n]);
}

// ---------------------------------------------------------------------------
// LayerNorm over d_model, emit bf16 activations; one token per block
// ---------------------------------------------------------------------------
__global__ __launch_bounds__(256) void ln_bf16_kernel(
    const float* __restrict__ x, const float* __restrict__ w,
    const float* __restrict__ b, unsigned short* __restrict__ hout)
{
  const int tok = blockIdx.x;
  const float* row = x + (size_t)tok * D_MODEL;
  float v[4];
  float s = 0.f, sq = 0.f;
#pragma unroll
  for (int i = 0; i < 4; ++i) {
    float t = row[threadIdx.x + i * 256];
    v[i] = t; s += t; sq += t * t;
  }
#pragma unroll
  for (int off = 16; off; off >>= 1) {
    s  += __shfl_xor(s,  off, 32);
    sq += __shfl_xor(sq, off, 32);
  }
  __shared__ float red[16];
  const int wave = threadIdx.x >> 5, lane = threadIdx.x & 31;
  if (lane == 0) { red[wave] = s; red[8 + wave] = sq; }
  __syncthreads();
  s = 0.f; sq = 0.f;
#pragma unroll
  for (int i = 0; i < 8; ++i) { s += red[i]; sq += red[8 + i]; }
  const float mean = s * (1.0f / D_MODEL);
  const float var  = sq * (1.0f / D_MODEL) - mean * mean;
  const float rstd = rsqrtf(var + LN_EPS);
#pragma unroll
  for (int i = 0; i < 4; ++i) {
    const int idx = threadIdx.x + i * 256;
    const float h = (v[i] - mean) * rstd * w[idx] + b[idx];
    hout[(size_t)tok * D_MODEL + idx] = f2bf(h);
  }
}

// ---------------------------------------------------------------------------
// bf16 WMMA GEMM, TDM double-buffered LDS staging
//   C[M,N] = A[M,K] (bf16 row-major) @ WT[N,K]^T (bf16)
// block = 8 waves; block tile 128M x 128N, K-step 64
// wave tile 32M x 64N: 2 M-subtiles x 4 N-tiles = 8 accumulators
// per stage: TDM loads A 128x64 (16KB) + B 128x64 (16KB); 16 WMMAs/wave
// EPI: 0 = store raw, 1 = softplus(acc + bias[n]), 2 = acc + resid[m,n]
// ---------------------------------------------------------------------------
template <int EPI>
__global__ __launch_bounds__(256) void gemm_wmma_bf16(
    const unsigned short* __restrict__ A,
    const unsigned short* __restrict__ WT,
    float* __restrict__ C,
    const float* __restrict__ aux,
    int M, int N, int K)
{
  __shared__ __align__(16) unsigned short sA[2][128 * 64];
  __shared__ __align__(16) unsigned short sB[2][128 * 64];

  const int lane  = threadIdx.x & 31;
  const int wave  = threadIdx.x >> 5;
  const int wm    = wave & 3;        // M sub-block of 32 rows
  const int wn    = wave >> 2;       // N sub-block of 64 cols
  const int blkM  = blockIdx.y * 128;
  const int blkN  = blockIdx.x * 128;
  const int hi    = lane >> 4;       // 0: k 0-7,16-23; 1: k 8-15,24-31 of each 32-k step
  const int l15   = lane & 15;

  const unsigned short* Ab = A  + (size_t)blkM * K;
  const unsigned short* Bb = WT + (size_t)blkN * K;
  const int nk = K >> 6;             // number of 64-wide K stages

  // prologue: fill buffer 0
  if (wave == 0) {
    tdm_load_tile_bf16((unsigned)(uintptr_t)&sA[0][0], Ab, 64u, 128u, (unsigned)K);
    tdm_load_tile_bf16((unsigned)(uintptr_t)&sB[0][0], Bb, 64u, 128u, (unsigned)K);
  }

  // LDS fragment offsets (elements); LDS row stride = 64
  const int arow0 = (wm * 32 + l15) * 64 + hi * 8;   // M-subtile 0
  const int arow1 = arow0 + 16 * 64;                 // M-subtile 1
  const int brow  = (wn * 64 + l15) * 64 + hi * 8;

  v8f acc[2][4] = {};

  for (int i = 0; i < nk; ++i) {
    const int cur = i & 1;
    __syncthreads();                       // all waves done reading buffer cur^1
    if (wave == 0) {
      if (i + 1 < nk) {
        const int nxt = cur ^ 1;
        tdm_load_tile_bf16((unsigned)(uintptr_t)&sA[nxt][0],
                           Ab + (size_t)(i + 1) * 64, 64u, 128u, (unsigned)K);
        tdm_load_tile_bf16((unsigned)(uintptr_t)&sB[nxt][0],
                           Bb + (size_t)(i + 1) * 64, 64u, 128u, (unsigned)K);
        __builtin_amdgcn_s_wait_tensorcnt(2);   // oldest 2 (buffer cur) done
      } else {
        __builtin_amdgcn_s_wait_tensorcnt(0);
      }
    }
    __syncthreads();                       // buffer cur visible to all waves

    const unsigned short* pA = &sA[cur][0];
    const unsigned short* pB = &sB[cur][0];
#pragma unroll
    for (int ks = 0; ks < 2; ++ks) {
      const int ko = ks * 32;
      // batch all fragment loads, then run 8 independent WMMAs back-to-back
      BfFrag a0, a1, bq[4];
      a0.h[0] = *(const us8*)(pA + arow0 + ko);
      a0.h[1] = *(const us8*)(pA + arow0 + ko + 16);
      a1.h[0] = *(const us8*)(pA + arow1 + ko);
      a1.h[1] = *(const us8*)(pA + arow1 + ko + 16);
#pragma unroll
      for (int j = 0; j < 4; ++j) {
        const int bo = brow + j * 16 * 64 + ko;
        bq[j].h[0] = *(const us8*)(pB + bo);
        bq[j].h[1] = *(const us8*)(pB + bo + 16);
      }
#pragma unroll
      for (int j = 0; j < 4; ++j) {
        acc[0][j] = __builtin_amdgcn_wmma_f32_16x16x32_bf16(
            false, a0.v, false, bq[j].v, (short)0, acc[0][j], false, false);
        acc[1][j] = __builtin_amdgcn_wmma_f32_16x16x32_bf16(
            false, a1.v, false, bq[j].v, (short)0, acc[1][j], false, false);
      }
    }
  }

  // C/D layout: VGPR r -> M = mrow + r, N = n  (rows 0-7 lanes<16, 8-15 lanes>=16)
#pragma unroll
  for (int mi = 0; mi < 2; ++mi) {
    const int mrow = blkM + wm * 32 + mi * 16 + hi * 8;
#pragma unroll
    for (int j = 0; j < 4; ++j) {
      const int n = blkN + wn * 64 + j * 16 + l15;
#pragma unroll
      for (int r = 0; r < 8; ++r) {
        float val = acc[mi][j][r];
        const size_t o = (size_t)(mrow + r) * N + n;
        if (EPI == 1)      val = softplusf_(val + aux[n]);
        else if (EPI == 2) val += aux[o];
        C[o] = val;
      }
    }
  }
}

// ---------------------------------------------------------------------------
// causal depthwise conv (k=4) + bias + SiLU; emits f32 and bf16 copies
// ---------------------------------------------------------------------------
__global__ __launch_bounds__(256) void conv_silu_kernel(
    const float* __restrict__ xz, const float* __restrict__ cw,
    const float* __restrict__ cb, float* __restrict__ xs,
    unsigned short* __restrict__ xsbf)
{
  const int tok = blockIdx.x;
  const int b = tok >> 10, t = tok & (SEQ_T - 1);
#pragma unroll
  for (int i = 0; i < 8; ++i) {
    const int c = threadIdx.x + i * 256;
    float acc = cb[c];
#pragma unroll
    for (int j = 0; j < D_CONV; ++j) {
      const int tt = t - (D_CONV - 1) + j;
      if (tt >= 0)
        acc += cw[c * D_CONV + j] *
               xz[(size_t)(b * SEQ_T + tt) * (2 * D_INNER) + c];
    }
    const float sv = siluf_(acc);
    const size_t o = (size_t)tok * D_INNER + c;
    xs[o]   = sv;
    xsbf[o] = f2bf(sv);
  }
}

// ---------------------------------------------------------------------------
// selective scan: one wave per (batch, channel); lanes split the 64 states
// ---------------------------------------------------------------------------
__global__ __launch_bounds__(256) void scan_kernel(
    const float* __restrict__ A_log, const float* __restrict__ dt,
    const float* __restrict__ xs, const float* __restrict__ btct,
    float* __restrict__ ysc)
{
  const int lane = threadIdx.x & 31;
  const int wid  = blockIdx.x * 8 + (threadIdx.x >> 5);   // 0..4095
  const int b    = wid >> 11;
  const int c    = wid & (D_INNER - 1);

  const float a0 = -__expf(A_log[(size_t)c * D_STATE + lane]);
  const float a1 = -__expf(A_log[(size_t)c * D_STATE + 32 + lane]);
  float s0 = 0.f, s1 = 0.f;

  for (int t = 0; t < SEQ_T; ++t) {
    const size_t tok = (size_t)b * SEQ_T + t;
    const float d    = dt[tok * D_INNER + c];     // wave-uniform
    const float xv   = xs[tok * D_INNER + c];     // wave-uniform
    const float* bc  = btct + tok * 128;          // [B(64) | C(64)]
    __builtin_prefetch((const void*)(bc + 128), 0, 0);  // next step's B/C row
    const float dx   = d * xv;
    s0 = s0 * __expf(d * a0) + dx * bc[lane];
    s1 = s1 * __expf(d * a1) + dx * bc[32 + lane];
    float p = s0 * bc[64 + lane] + s1 * bc[96 + lane];
#pragma unroll
    for (int off = 16; off; off >>= 1) p += __shfl_xor(p, off, 32);
    if (lane == 0) ysc[tok * D_INNER + c] = p;
  }
}

// ---------------------------------------------------------------------------
// y = (y_scan + xs * D[c]) * silu(z); emit bf16 for the output GEMM
// ---------------------------------------------------------------------------
__global__ __launch_bounds__(256) void gate_kernel(
    const float* __restrict__ ysc, const float* __restrict__ xs,
    const float* __restrict__ Dp, const float* __restrict__ xz,
    unsigned short* __restrict__ ygbf)
{
  const size_t total = (size_t)NTOK * D_INNER;
  const size_t gid = (size_t)blockIdx.x * 256 + threadIdx.x;
  if (gid >= total) return;
  const size_t tok = gid >> 11;
  const int c = (int)(gid & (D_INNER - 1));
  const float y = ysc[gid] + xs[gid] * Dp[c];
  const float z = xz[tok * (2 * D_INNER) + D_INNER + c];
  ygbf[gid] = f2bf(y * siluf_(z));
}

// ---------------------------------------------------------------------------
extern "C" void kernel_launch(void* const* d_in, const int* in_sizes, int n_in,
                              void* d_out, int out_size, void* d_ws, size_t ws_size,
                              hipStream_t stream)
{
  (void)in_sizes; (void)n_in; (void)out_size; (void)ws_size;

  const float* x      = (const float*)d_in[0];
  const float* norm_w = (const float*)d_in[1];
  const float* norm_b = (const float*)d_in[2];
  const float* Win    = (const float*)d_in[3];
  const float* conv_w = (const float*)d_in[4];
  const float* conv_b = (const float*)d_in[5];
  const float* dt_w   = (const float*)d_in[6];
  const float* dt_b   = (const float*)d_in[7];
  const float* A_log  = (const float*)d_in[8];
  const float* Dp     = (const float*)d_in[9];
  const float* Bp     = (const float*)d_in[10];
  const float* Cp     = (const float*)d_in[11];
  const float* Wout   = (const float*)d_in[12];
  float* out = (float*)d_out;

  char* wsb = (char*)d_ws;
  size_t off = 0;
  auto alloc = [&](size_t bytes) -> void* {
    void* p = wsb + off;
    off = (off + bytes + 255) & ~(size_t)255;
    return p;
  };

  unsigned short* WinT  = (unsigned short*)alloc(sizeof(unsigned short) * (size_t)4096 * 1024);
  unsigned short* dtwT  = (unsigned short*)alloc(sizeof(unsigned short) * (size_t)2048 * 2048);
  unsigned short* WoutT = (unsigned short*)alloc(sizeof(unsigned short) * (size_t)1024 * 2048);
  unsigned short* BCT   = (unsigned short*)alloc(sizeof(unsigned short) * (size_t)128  * 2048);
  unsigned short* hbf   = (unsigned short*)alloc(sizeof(unsigned short) * (size_t)NTOK * D_MODEL);
  float*          xzf   = (float*)alloc(sizeof(float) * (size_t)NTOK * 2 * D_INNER);
  float*          xsf   = (float*)alloc(sizeof(float) * (size_t)NTOK * D_INNER);
  unsigned short* xsbf  = (unsigned short*)alloc(sizeof(unsigned short) * (size_t)NTOK * D_INNER);
  float*          dtf   = (float*)alloc(sizeof(float) * (size_t)NTOK * D_INNER);
  float*          btct  = (float*)alloc(sizeof(float) * (size_t)NTOK * 128);
  float*          ysc   = (float*)alloc(sizeof(float) * (size_t)NTOK * D_INNER);
  unsigned short* ygbf  = (unsigned short*)alloc(sizeof(unsigned short) * (size_t)NTOK * D_INNER);

  auto nblk = [](size_t n) { return dim3((unsigned)((n + 255) / 256)); };

  // weight bf16 transposes
  cvt_transpose_kernel<<<nblk((size_t)1024 * 4096), 256, 0, stream>>>(Win,  WinT,  1024, 4096);
  cvt_transpose_kernel<<<nblk((size_t)2048 * 2048), 256, 0, stream>>>(dt_w, dtwT,  2048, 2048);
  cvt_transpose_kernel<<<nblk((size_t)2048 * 1024), 256, 0, stream>>>(Wout, WoutT, 2048, 1024);
  cvt_transpose_kernel<<<nblk((size_t)2048 * 64),   256, 0, stream>>>(Bp,   BCT,             2048, 64);
  cvt_transpose_kernel<<<nblk((size_t)2048 * 64),   256, 0, stream>>>(Cp,   BCT + 64 * 2048, 2048, 64);

  // 1) LayerNorm -> bf16
  ln_bf16_kernel<<<NTOK, 256, 0, stream>>>(x, norm_w, norm_b, hbf);

  // 2) xz = h @ Win   [2048 x 4096]
  gemm_wmma_bf16<0><<<dim3(4096 / 128, NTOK / 128), 256, 0, stream>>>(
      hbf, WinT, xzf, nullptr, NTOK, 4096, 1024);

  // 3) depthwise causal conv + SiLU
  conv_silu_kernel<<<NTOK, 256, 0, stream>>>(xzf, conv_w, conv_b, xsf, xsbf);

  // 4) dt = softplus(xs @ dt_w + dt_b)   [2048 x 2048]
  gemm_wmma_bf16<1><<<dim3(2048 / 128, NTOK / 128), 256, 0, stream>>>(
      xsbf, dtwT, dtf, dt_b, NTOK, 2048, 2048);

  // 5) [Bt | Ct] = xs @ [Bp | Cp]   [2048 x 128]
  gemm_wmma_bf16<0><<<dim3(1, NTOK / 128), 256, 0, stream>>>(
      xsbf, BCT, btct, nullptr, NTOK, 128, 2048);

  // 6) selective scan (4096 waves)
  scan_kernel<<<512, 256, 0, stream>>>(A_log, dtf, xsf, btct, ysc);

  // 7) gating
  gate_kernel<<<nblk((size_t)NTOK * D_INNER), 256, 0, stream>>>(ysc, xsf, Dp, xzf, ygbf);

  // 8) out = x + y @ Wout   [2048 x 1024]
  gemm_wmma_bf16<2><<<dim3(1024 / 128, NTOK / 128), 256, 0, stream>>>(
      ygbf, WoutT, out, x, NTOK, 1024, 2048);
}
